// BlaschkeNetwork1d_52931176956390
// MI455X (gfx1250) — matrix operand order repaired and checked by reading, not compiled, not run on hardware
//
#include <hip/hip_runtime.h>
#include <math.h>

// ---------------------------------------------------------------------------
// Types for CDNA5 WMMA (wave32): v_wmma_f32_16x16x32_bf16
// ---------------------------------------------------------------------------
typedef __bf16 bf16;
typedef __bf16 v16bf __attribute__((ext_vector_type(16)));
typedef __bf16 v8bf  __attribute__((ext_vector_type(8)));
typedef float  v8f   __attribute__((ext_vector_type(8)));

#define B_      64
#define C_      4
#define L_      2048
#define NP_     128
#define D_      128
#define DFF_    512
#define NC_     26
#define LAYERS_ 6
#define NF_     624          // LAYERS*C*NC
#define TOK_    65536        // (B*C*2) * NP tokens through the transformer

// ---------------------------------------------------------------------------
// WMMA helpers
// ---------------------------------------------------------------------------
__device__ __forceinline__ v8f wmma_bf16(v16bf a, v16bf b, v8f c) {
    // 8 args: (neg_a, A, neg_b, B, c_mod, C, reuse_a, reuse_b)
    return __builtin_amdgcn_wmma_f32_16x16x32_bf16(false, a, false, b,
                                                   (short)0, c, false, false);
}

// Load the A fragment for one lane. `rowPtr` must already point at
// A + row*lda + k0 + hsel*16, where hsel = lane>>4.
// Contiguous k order: lane<16: elems 0..15 = K[k0..k0+15]
//                     lane>=16: K[k0+16..k0+31] (handled by the hsel*16 offset).
__device__ __forceinline__ v16bf load_afrag(const bf16* rowPtr) {
    const v8bf* pa = (const v8bf*)rowPtr;
    v8bf lo = pa[0];
    v8bf hi = pa[1];           // +8 elements
    v16bf r;
#pragma unroll
    for (int i = 0; i < 8; ++i) { r[i] = lo[i]; r[8 + i] = hi[i]; }
    return r;
}

__device__ __forceinline__ float gelu_tanh(float x) {
    float x3 = x * x * x;
    return 0.5f * x * (1.0f + tanhf(0.7978845608028654f * (x + 0.044715f * x3)));
}

// ---------------------------------------------------------------------------
// Generic bf16 WMMA GEMM:  C[M,N] = A[M,K](bf16,row-major) * B  (+bias, +act)
// B is given transposed: BT[N,K] bf16 row-major -> each B fragment is one
// contiguous 32-byte load per lane.
// Wave tile: 16(M) x 64(N); workgroup = 8 waves stacked in M (128 x 64 tile).
// Requires: M % 128 == 0, N % 64 == 0, K % 32 == 0.
// ACT: 0 = none, 1 = tanh-GELU.  OUTBF: 0 -> f32 Cf, 1 -> bf16 Cb.
// ---------------------------------------------------------------------------
template <int ACT, int OUTBF>
__global__ __launch_bounds__(256) void gemm_wmma(
    const bf16* __restrict__ A, int lda,
    const bf16* __restrict__ BT, int ldb,
    const float* __restrict__ bias,
    float* __restrict__ Cf, bf16* __restrict__ Cb, int ldc, int K)
{
    const int lane = threadIdx.x & 31;
    const int wave = threadIdx.x >> 5;
    const int l15  = lane & 15;
    const int hsel = lane >> 4;
    const int m0   = blockIdx.x * 128 + wave * 16;
    const int n0   = blockIdx.y * 64;

    const bf16* arow = A + (size_t)(m0 + l15) * lda + hsel * 16;

    v8f acc0 = {}, acc1 = {}, acc2 = {}, acc3 = {};
    for (int k0 = 0; k0 < K; k0 += 32) {
        // speculative prefetch of next A fragment (global_prefetch_b8);
        // tail over-read is dropped by HW per ISA prefetch rules.
        __builtin_prefetch(arow + k0 + 32, 0, 1);
        v16bf af = load_afrag(arow + k0);
        const bf16* bbase = BT + (size_t)(n0 + l15) * ldb + k0 + hsel * 16;
        v16bf b0 = *(const v16bf*)(bbase);
        v16bf b1 = *(const v16bf*)(bbase + (size_t)16 * ldb);
        v16bf b2 = *(const v16bf*)(bbase + (size_t)32 * ldb);
        v16bf b3 = *(const v16bf*)(bbase + (size_t)48 * ldb);
        acc0 = wmma_bf16(af, b0, acc0);
        acc1 = wmma_bf16(af, b1, acc1);
        acc2 = wmma_bf16(af, b2, acc2);
        acc3 = wmma_bf16(af, b3, acc3);
    }

    // C layout: VGPR j -> M = m0 + j + hsel*8, N = n0 + t*16 + l15
    const int mb = m0 + hsel * 8;
    v8f accs[4] = {acc0, acc1, acc2, acc3};
#pragma unroll
    for (int t = 0; t < 4; ++t) {
        int n = n0 + t * 16 + l15;
        float bv = bias ? bias[n] : 0.0f;
#pragma unroll
        for (int j = 0; j < 8; ++j) {
            float x = accs[t][j] + bv;
            if (ACT == 1) x = gelu_tanh(x);
            size_t idx = (size_t)(mb + j) * ldc + n;
            if (OUTBF) Cb[idx] = (bf16)x;
            else       Cf[idx] = x;
        }
    }
}

// ---------------------------------------------------------------------------
// Fused attention: one workgroup per (sequence, head). P=128 tokens, dh=32.
// Everything stays in LDS (56 KB): scores via WMMA, wave-parallel softmax,
// P*V via WMMA reading fragments from LDS.
// ---------------------------------------------------------------------------
__global__ __launch_bounds__(256) void attn_kernel(
    const bf16* __restrict__ qbf, const bf16* __restrict__ kbf,
    const bf16* __restrict__ vbf, bf16* __restrict__ obf)
{
    __shared__ __attribute__((aligned(32))) bf16 Qs[128 * 32];
    __shared__ __attribute__((aligned(32))) bf16 Ks[128 * 32];
    __shared__ __attribute__((aligned(32))) bf16 Vt[32 * 128];   // transposed V
    __shared__ __attribute__((aligned(32))) bf16 Sb[128 * 128];  // scores/probs

    const int seq = blockIdx.x >> 2;
    const int hh  = blockIdx.x & 3;
    const int tid = threadIdx.x;

    {   // stage q,k,v for this head into LDS (v transposed)
        int p  = tid >> 1;
        int ho = (tid & 1) * 16;
        size_t base = ((size_t)(seq * 128 + p)) * D_ + hh * 32 + ho;
#pragma unroll
        for (int i = 0; i < 16; ++i) {
            Qs[p * 32 + ho + i] = qbf[base + i];
            Ks[p * 32 + ho + i] = kbf[base + i];
            Vt[(ho + i) * 128 + p] = vbf[base + i];
        }
    }
    __syncthreads();

    const int lane = tid & 31, wave = tid >> 5;
    const int l15 = lane & 15, hsel = lane >> 4;

    {   // scores: wave w owns M-tile w, sweeps all 8 N-tiles. K = dh = 32.
        v16bf af = load_afrag(Qs + (wave * 16 + l15) * 32 + hsel * 16);
        const float sc = 0.17677669529663687f;   // 1/sqrt(32)
        const int mb = wave * 16 + hsel * 8;
#pragma unroll
        for (int tj = 0; tj < 8; ++tj) {
            v16bf bf_ = *(const v16bf*)(Ks + (tj * 16 + l15) * 32 + hsel * 16);
            v8f acc = {};
            acc = wmma_bf16(af, bf_, acc);
            int n = tj * 16 + l15;
#pragma unroll
            for (int j = 0; j < 8; ++j)
                Sb[(mb + j) * 128 + n] = (bf16)(acc[j] * sc);
        }
    }
    __syncthreads();

    // wave-parallel row softmax: wave w owns rows {w, w+8, ...};
    // lanes cover contiguous columns (conflict-free LDS), reduce via shfl_xor.
    for (int row = wave; row < 128; row += 8) {
        bf16* rp = Sb + row * 128;
        float x0 = (float)rp[lane];
        float x1 = (float)rp[lane + 32];
        float x2 = (float)rp[lane + 64];
        float x3 = (float)rp[lane + 96];
        float mx = fmaxf(fmaxf(x0, x1), fmaxf(x2, x3));
#pragma unroll
        for (int off = 16; off > 0; off >>= 1)
            mx = fmaxf(mx, __shfl_xor(mx, off, 32));
        float e0 = __expf(x0 - mx), e1 = __expf(x1 - mx);
        float e2 = __expf(x2 - mx), e3 = __expf(x3 - mx);
        float s = e0 + e1 + e2 + e3;
#pragma unroll
        for (int off = 16; off > 0; off >>= 1)
            s += __shfl_xor(s, off, 32);
        float inv = 1.0f / s;
        rp[lane]      = (bf16)(e0 * inv);
        rp[lane + 32] = (bf16)(e1 * inv);
        rp[lane + 64] = (bf16)(e2 * inv);
        rp[lane + 96] = (bf16)(e3 * inv);
    }
    __syncthreads();

    // O = P(128x128) * V(128x32): wave w owns M-tile w; 2 N-tiles of dh.
#pragma unroll
    for (int nt = 0; nt < 2; ++nt) {
        v8f acc = {};
        for (int k0 = 0; k0 < 128; k0 += 32) {
            v16bf af = load_afrag(Sb + (wave * 16 + l15) * 128 + k0 + hsel * 16);
            v16bf bf_ = *(const v16bf*)(Vt + (nt * 16 + l15) * 128 + k0 + hsel * 16);
            acc = wmma_bf16(af, bf_, acc);
        }
        int dh = nt * 16 + l15;
        int mb = wave * 16 + hsel * 8;
#pragma unroll
        for (int j = 0; j < 8; ++j)
            obf[((size_t)(seq * 128 + mb + j)) * D_ + hh * 32 + dh] = (bf16)acc[j];
    }
}

// ---------------------------------------------------------------------------
// Small kernels
// ---------------------------------------------------------------------------
__global__ void transposeW_kernel(const float* __restrict__ W, bf16* __restrict__ WT,
                                  int K, int N) {
    int idx = blockIdx.x * blockDim.x + threadIdx.x;
    if (idx >= K * N) return;
    int k = idx / N, n = idx % N;
    WT[(size_t)n * K + k] = (bf16)W[(size_t)k * N + n];
}

// circular Hilbert/analytic kernel: K[d] = (1/L) sum_k h[k] e^{+2pi i k d / L}
__global__ void hilbert_build(float* __restrict__ Kr, float* __restrict__ Ki) {
    int d = blockIdx.x * blockDim.x + threadIdx.x;
    if (d >= L_) return;
    float sr = 1.0f + ((d & 1) ? -1.0f : 1.0f);   // k=0 and k=L/2 terms
    float si = 0.0f;
    for (int k = 1; k < L_ / 2; ++k) {
        float ang = 6.283185307179586f * (float)((k * d) & (L_ - 1)) / (float)L_;
        sr += 2.0f * cosf(ang);
        si += 2.0f * sinf(ang);
    }
    Kr[d] = sr / (float)L_;
    Ki[d] = si / (float)L_;
}

__global__ void analytic_kernel(const float* __restrict__ x,
                                const float* __restrict__ Kr, const float* __restrict__ Ki,
                                float* __restrict__ rre, float* __restrict__ rim) {
    __shared__ float xs[L_];
    __shared__ float KrS[L_];
    __shared__ float KiS[L_];
    int bc = blockIdx.x;
    for (int i = threadIdx.x; i < L_; i += blockDim.x) {
        xs[i]  = x[(size_t)bc * L_ + i];
        KrS[i] = Kr[i];
        KiS[i] = Ki[i];
    }
    __syncthreads();
    for (int n = threadIdx.x; n < L_; n += blockDim.x) {
        float ar = 0.0f, ai = 0.0f;
        for (int m = 0; m < L_; ++m) {
            int d = (n - m) & (L_ - 1);
            float xv = xs[m];
            ar += xv * KrS[d];
            ai += xv * KiS[d];
        }
        rre[(size_t)bc * L_ + n] = ar;
        rim[(size_t)bc * L_ + n] = ai;
    }
}

// patch embedding: h[t,d] = sum_s x[n,v,p,s]*ew[s,d] + eb[d] + pos[p,d]
__global__ void embed_kernel(const float* __restrict__ rre, const float* __restrict__ rim,
                             const float* __restrict__ ew, const float* __restrict__ eb,
                             const float* __restrict__ pos,
                             float* __restrict__ h, bf16* __restrict__ hbf) {
    size_t idx = (size_t)blockIdx.x * blockDim.x + threadIdx.x;
    if (idx >= (size_t)TOK_ * D_) return;
    int d = (int)(idx & 127);
    int t = (int)(idx >> 7);
    int p = t & 127;
    int v = (t >> 7) & 1;
    int n = t >> 8;
    const float* xsrc = (v ? rim : rre) + (size_t)n * L_ + p * 16;
    float acc = eb[d] + pos[p * D_ + d];
#pragma unroll
    for (int s = 0; s < 16; ++s) acc += xsrc[s] * ew[s * D_ + d];
    h[idx]   = acc;
    hbf[idx] = (bf16)acc;
}

// h = tanh(a*(h + C))*w + b  ; also refresh bf16 copy
__global__ void dyt_kernel(float* __restrict__ h, bf16* __restrict__ hbf,
                           const float* __restrict__ Cin,
                           const float* __restrict__ aP,
                           const float* __restrict__ w, const float* __restrict__ bb) {
    size_t idx = (size_t)blockIdx.x * blockDim.x + threadIdx.x;
    if (idx >= (size_t)TOK_ * D_) return;
    int d = (int)(idx & 127);
    float x = h[idx] + Cin[idx];
    float y = tanhf(aP[0] * x) * w[d] + bb[d];
    h[idx]   = y;
    hbf[idx] = (bf16)y;
}

// head: p[n, j] = head_b[j] + sum_d hcat[n,d] * head_w[d,j]   (d over 2*D=256)
__global__ void head_kernel(const float* __restrict__ h,
                            const float* __restrict__ hw, const float* __restrict__ hb,
                            float* __restrict__ pbuf) {
    int n = blockIdx.x;
    int j = threadIdx.x;
    if (j >= NC_) return;
    float acc = hb[j];
    for (int d = 0; d < 2 * D_; ++d) {
        int v = d >> 7, dd = d & 127;
        size_t tok = ((size_t)(n * 2 + v)) * NP_ + (NP_ - 1);
        acc += h[tok * D_ + dd] * hw[d * NC_ + j];
    }
    pbuf[n * NC_ + j] = acc;
}

// transform raw params -> (alpha, softplus(beta+eps), sigmoid(gamma), sr, si)
__global__ void paramfix_kernel(const float* __restrict__ pbuf,
                                float* __restrict__ feats, int layer) {
    int n = blockIdx.x;
    int j = threadIdx.x;
    if (j >= NC_) return;
    float x = pbuf[n * NC_ + j];
    float y;
    if (j < 8)       y = x;
    else if (j < 16) { float t = x + 1e-6f; y = (t > 20.0f) ? t : log1pf(expf(t)); }
    else if (j < 24) y = 1.0f / (1.0f + expf(-x));
    else             y = x;
    int b = n >> 2, c = n & 3;
    feats[(((size_t)b * LAYERS_ + layer) * C_ + c) * NC_ + j] = y;
}

// residual = residual * conj(exp(i*phase)) - (sr + i*si)
__global__ void blaschke_kernel(float* __restrict__ rre, float* __restrict__ rim,
                                const float* __restrict__ feats, int layer) {
    __shared__ float prm[NC_];
    int bc = blockIdx.x >> 3;
    int chunk = blockIdx.x & 7;
    int b = bc >> 2, c = bc & 3;
    if (threadIdx.x < NC_)
        prm[threadIdx.x] =
            feats[(((size_t)b * LAYERS_ + layer) * C_ + c) * NC_ + threadIdx.x];
    __syncthreads();
    int l = chunk * 256 + threadIdx.x;
    size_t i = (size_t)bc * L_ + l;
    float re = rre[i], im = rim[i];
    float phase = 0.0f;
#pragma unroll
    for (int r = 0; r < 8; ++r)
        phase += prm[16 + r] * (atanf((re - prm[r]) / prm[8 + r]) + 1.5707963267948966f);
    float sp, cp;
    sincosf(phase, &sp, &cp);
    rre[i] = re * cp + im * sp - prm[24];
    rim[i] = im * cp - re * sp - prm[25];
}

__global__ void bn_kernel(const float* __restrict__ in, float* __restrict__ out,
                          const float* __restrict__ g, const float* __restrict__ b,
                          const float* __restrict__ m, const float* __restrict__ v,
                          int total) {
    int idx = blockIdx.x * blockDim.x + threadIdx.x;
    if (idx >= total) return;
    int d = idx % NF_;
    out[idx] = (in[idx] - m[d]) * rsqrtf(v[d] + 1e-5f) * g[d] + b[d];
}

__global__ void fc1_kernel(const float* __restrict__ fn, const float* __restrict__ w,
                           const float* __restrict__ bias, float* __restrict__ z) {
    int idx = blockIdx.x * blockDim.x + threadIdx.x;
    if (idx >= B_ * NF_) return;
    int n = idx / NF_, j = idx % NF_;
    float acc = bias[j];
    const float* fr = fn + (size_t)n * NF_;
    for (int d = 0; d < NF_; ++d) acc += fr[d] * w[(size_t)d * NF_ + j];
    z[idx] = acc / (1.0f + expf(-acc));   // silu
}

__global__ void fc2_kernel(const float* __restrict__ z2, const float* __restrict__ w,
                           const float* __restrict__ bias, float* __restrict__ out) {
    int idx = blockIdx.x * blockDim.x + threadIdx.x;
    if (idx >= B_ * 10) return;
    int n = idx / 10, o = idx % 10;
    float acc = bias[o];
    for (int j = 0; j < NF_; ++j) acc += z2[(size_t)n * NF_ + j] * w[(size_t)j * 10 + o];
    out[idx] = acc;
}

// ---------------------------------------------------------------------------
// Host: input map assumes setup_inputs() dict insertion order:
//   0:x 1:embed_w 2:embed_b 3:pos
//   blocks (3x, base=4+18*blk): +0 Wq +1 bq +2 Wk +3 bk +4 Wv +5 bv +6 Wo +7 bo
//     +8 dyt1_a +9 dyt1_w +10 dyt1_b +11 W1 +12 b1 +13 W2 +14 b2
//     +15 dyt2_a +16 dyt2_w +17 dyt2_b
//   58:head_w 59:head_b 60..63:bn1_{g,b,m,v} 64:fc1_w 65:fc1_b
//   66..69:bn2_{g,b,m,v} 70:fc2_w 71:fc2_b
// Workspace required: ~150 MB.
// ---------------------------------------------------------------------------
extern "C" void kernel_launch(void* const* d_in, const int* in_sizes, int n_in,
                              void* d_out, int out_size, void* d_ws, size_t ws_size,
                              hipStream_t stream)
{
    (void)in_sizes; (void)n_in; (void)out_size; (void)ws_size;
    auto F = [&](int i) -> const float* { return (const float*)d_in[i]; };

    const float* x       = F(0);
    const float* embed_w = F(1);
    const float* embed_b = F(2);
    const float* pos     = F(3);
    const float* head_w  = F(58);
    const float* head_b  = F(59);
    const float* bn1_g = F(60), *bn1_b = F(61), *bn1_m = F(62), *bn1_v = F(63);
    const float* fc1_w = F(64), *fc1_b = F(65);
    const float* bn2_g = F(66), *bn2_b = F(67), *bn2_m = F(68), *bn2_v = F(69);
    const float* fc2_w = F(70), *fc2_b = F(71);

    char* wsp = (char*)d_ws;
    auto alloc = [&](size_t bytes) -> char* {
        char* p = wsp;
        wsp += (bytes + 255) & ~(size_t)255;
        return p;
    };

    float* Kr  = (float*)alloc(L_ * 4);
    float* Ki  = (float*)alloc(L_ * 4);
    float* rre = (float*)alloc((size_t)B_ * C_ * L_ * 4);
    float* rim = (float*)alloc((size_t)B_ * C_ * L_ * 4);
    float* h   = (float*)alloc((size_t)TOK_ * D_ * 4);
    bf16*  hbf = (bf16*)alloc((size_t)TOK_ * D_ * 2);
    char*  bufA = alloc((size_t)TOK_ * DFF_ * 2);        // 64 MB, aliased
    bf16* qbf  = (bf16*)bufA;
    bf16* kbf  = (bf16*)(bufA + (size_t)TOK_ * D_ * 2);
    bf16* vbf  = (bf16*)(bufA + (size_t)TOK_ * D_ * 4);
    bf16* obf  = (bf16*)(bufA + (size_t)TOK_ * D_ * 6);
    bf16* ffbf = (bf16*)bufA;                            // FFN reuses attention space
    float* Cf  = (float*)alloc((size_t)TOK_ * D_ * 4);

    bf16* wT[3][6];
    for (int blk = 0; blk < 3; ++blk) {
        for (int i = 0; i < 4; ++i) wT[blk][i] = (bf16*)alloc(D_ * D_ * 2);
        wT[blk][4] = (bf16*)alloc(D_ * DFF_ * 2);  // W1^T (512x128)
        wT[blk][5] = (bf16*)alloc(D_ * DFF_ * 2);  // W2^T (128x512)
    }
    float* pbuf  = (float*)alloc(256 * NC_ * 4);
    float* feats = (float*)alloc((size_t)B_ * NF_ * 4);
    float* fn    = (float*)alloc((size_t)B_ * NF_ * 4);
    float* z1    = (float*)alloc((size_t)B_ * NF_ * 4);
    float* z2    = (float*)alloc((size_t)B_ * NF_ * 4);

    // ---- one-time per launch: weight transposes + analytic signal ----------
    for (int blk = 0; blk < 3; ++blk) {
        int base = 4 + 18 * blk;
        transposeW_kernel<<<(D_ * D_ + 255) / 256, 256, 0, stream>>>(F(base + 0), wT[blk][0], D_, D_);
        transposeW_kernel<<<(D_ * D_ + 255) / 256, 256, 0, stream>>>(F(base + 2), wT[blk][1], D_, D_);
        transposeW_kernel<<<(D_ * D_ + 255) / 256, 256, 0, stream>>>(F(base + 4), wT[blk][2], D_, D_);
        transposeW_kernel<<<(D_ * D_ + 255) / 256, 256, 0, stream>>>(F(base + 6), wT[blk][3], D_, D_);
        transposeW_kernel<<<(D_ * DFF_ + 255) / 256, 256, 0, stream>>>(F(base + 11), wT[blk][4], D_, DFF_);
        transposeW_kernel<<<(D_ * DFF_ + 255) / 256, 256, 0, stream>>>(F(base + 13), wT[blk][5], DFF_, D_);
    }
    hilbert_build<<<L_ / 256, 256, 0, stream>>>(Kr, Ki);
    analytic_kernel<<<B_ * C_, 256, 0, stream>>>(x, Kr, Ki, rre, rim);

    const dim3 gP(TOK_ / 128, D_ / 64);      // (512, 2)
    const dim3 gFF(TOK_ / 128, DFF_ / 64);   // (512, 8)
    const int  elemBlocks = (TOK_ * D_) / 256;

    for (int layer = 0; layer < LAYERS_; ++layer) {
        embed_kernel<<<elemBlocks, 256, 0, stream>>>(rre, rim, embed_w, embed_b, pos, h, hbf);

        for (int blk = 0; blk < 3; ++blk) {
            int base = 4 + 18 * blk;
            // QKV projections (bf16 out)
            gemm_wmma<0, 1><<<gP, 256, 0, stream>>>(hbf, D_, wT[blk][0], D_, F(base + 1), nullptr, qbf, D_, D_);
            gemm_wmma<0, 1><<<gP, 256, 0, stream>>>(hbf, D_, wT[blk][1], D_, F(base + 3), nullptr, kbf, D_, D_);
            gemm_wmma<0, 1><<<gP, 256, 0, stream>>>(hbf, D_, wT[blk][2], D_, F(base + 5), nullptr, vbf, D_, D_);
            // fused attention per (seq, head)
            attn_kernel<<<512 * 4, 256, 0, stream>>>(qbf, kbf, vbf, obf);
            // output projection (f32 out) + residual + DyT1
            gemm_wmma<0, 0><<<gP, 256, 0, stream>>>(obf, D_, wT[blk][3], D_, F(base + 7), Cf, nullptr, D_, D_);
            dyt_kernel<<<elemBlocks, 256, 0, stream>>>(h, hbf, Cf, F(base + 8), F(base + 9), F(base + 10));
            // FFN: GELU(h@W1+b1)@W2+b2, residual + DyT2
            gemm_wmma<1, 1><<<gFF, 256, 0, stream>>>(hbf, D_, wT[blk][4], D_, F(base + 12), nullptr, ffbf, DFF_, D_);
            gemm_wmma<0, 0><<<gP, 256, 0, stream>>>(ffbf, DFF_, wT[blk][5], DFF_, F(base + 14), Cf, nullptr, D_, DFF_);
            dyt_kernel<<<elemBlocks, 256, 0, stream>>>(h, hbf, Cf, F(base + 15), F(base + 16), F(base + 17));
        }

        head_kernel<<<256, 32, 0, stream>>>(h, head_w, head_b, pbuf);
        paramfix_kernel<<<256, 32, 0, stream>>>(pbuf, feats, layer);
        blaschke_kernel<<<256 * 8, 256, 0, stream>>>(rre, rim, feats, layer);
    }

    // final MLP head
    bn_kernel<<<(B_ * NF_ + 255) / 256, 256, 0, stream>>>(feats, fn, bn1_g, bn1_b, bn1_m, bn1_v, B_ * NF_);
    fc1_kernel<<<(B_ * NF_ + 255) / 256, 256, 0, stream>>>(fn, fc1_w, fc1_b, z1);
    bn_kernel<<<(B_ * NF_ + 255) / 256, 256, 0, stream>>>(z1, z2, bn2_g, bn2_b, bn2_m, bn2_v, B_ * NF_);
    fc2_kernel<<<(B_ * 10 + 255) / 256, 256, 0, stream>>>(z2, fc2_w, fc2_b, (float*)d_out);
}
